// Decoder_65335042507067
// MI455X (gfx1250) — matrix operand order repaired and checked by reading, not compiled, not run on hardware
//
#include <hip/hip_runtime.h>
#include <hip/hip_bf16.h>

typedef __attribute__((ext_vector_type(16))) __bf16 v16bf;
typedef __attribute__((ext_vector_type(8)))  __bf16 v8bf;
typedef __attribute__((ext_vector_type(8)))  float  v8f;

#define Tdim 1024
#define Sdim 4096
#define Hdim 1024
#define Gdim 4096   // 4*H

// ---------- helpers ----------
__device__ __forceinline__ unsigned short f2bf(float f) {
    unsigned int u = __float_as_uint(f);
    unsigned int r = u + 0x7FFFu + ((u >> 16) & 1u);
    return (unsigned short)(r >> 16);
}
__device__ __forceinline__ float bf2f(unsigned int u16) {
    return __uint_as_float(u16 << 16);
}
__device__ __forceinline__ float sigm(float x) {
    return 1.0f / (1.0f + __expf(-x));
}

// ---------- f32 -> bf16 conversion ----------
__global__ void cvt_bf16_k(const float* __restrict__ src, unsigned short* __restrict__ dst, int n) {
    int i = blockIdx.x * blockDim.x + threadIdx.x;
    int stride = gridDim.x * blockDim.x;
    for (; i < n; i += stride) dst[i] = f2bf(src[i]);
}

// ---------- generic f32 copy ----------
__global__ void copy_f32_k(float* __restrict__ dst, const float* __restrict__ src, int n) {
    int i = blockIdx.x * blockDim.x + threadIdx.x;
    int stride = gridDim.x * blockDim.x;
    for (; i < n; i += stride) dst[i] = src[i];
}

// ---------- WMMA bf16 GEMM:  C[M,N] = A[M,K] @ W[N,K]^T + b0[N] + b1[N] ----------
// One 16x16 output tile per wave. v_wmma_f32_16x16x32_bf16, f32 accumulation.
__global__ void gemm_wmma_bf16_k(const unsigned short* __restrict__ A,
                                 const unsigned short* __restrict__ W,
                                 const float* __restrict__ b0,
                                 const float* __restrict__ b1,
                                 float* __restrict__ C,
                                 int M, int N, int K) {
    int wid  = (blockIdx.x * blockDim.x + threadIdx.x) >> 5;
    int lane = threadIdx.x & 31;
    int tilesN = N >> 4;
    int tm = wid / tilesN;
    int tn = wid - tm * tilesN;
    if (tm >= (M >> 4)) return;

    int half = lane >> 4;    // 0: lanes 0-15, 1: lanes 16-31
    int l15  = lane & 15;

    // A: lane holds one row (M = tile_m*16 + l15); half selects a contiguous K half (k = 16*half + s).
    const unsigned short* arow = A + (size_t)(tm * 16 + l15) * K;
    // B = W^T: lane holds one column n -> contiguous row n of W; half selects K half.
    const unsigned short* brow = W + (size_t)(tn * 16 + l15) * K;

    v8f acc = {0.f, 0.f, 0.f, 0.f, 0.f, 0.f, 0.f, 0.f};

    for (int k0 = 0; k0 < K; k0 += 32) {
        v16bf a = *(const v16bf*)(arow + k0 + half * 16);       // K = k0 + half*16 .. +16
        v16bf b = *(const v16bf*)(brow + k0 + half * 16);       // K = k0 + half*16 .. +16
        acc = __builtin_amdgcn_wmma_f32_16x16x32_bf16(
            /*neg_a=*/false, a, /*neg_b=*/false, b,
            /*c_mod=*/(short)0, acc, /*reuse_a=*/false, /*reuse_b=*/false);
    }

    int n = tn * 16 + l15;
    float bias = b0[n] + b1[n];
#pragma unroll
    for (int j = 0; j < 8; ++j) {
        int row = tm * 16 + j + half * 8;   // C/D layout: VGPR j holds M = j (+8 for hi half)
        C[(size_t)row * N + n] = acc[j] + bias;
    }
}

// ---------- one LSTM timestep ----------
// grid = 32 blocks x 256 threads; block b owns hidden units [b*32, b*32+32).
// gates = xg[t] + h_in @ w_hh^T ; then fused pointwise update.
__global__ void lstm_step_k(const float* __restrict__ xg,
                            const unsigned short* __restrict__ w_hh,
                            const float* __restrict__ h_in,
                            float* __restrict__ h_out,
                            float* __restrict__ c,
                            float* __restrict__ out_f32,        // may be null
                            unsigned short* __restrict__ out_bf, // may be null
                            int t) {
    __shared__ float hS[Hdim];
    __shared__ float gS[128];

    int tid  = threadIdx.x;
    int wave = tid >> 5;
    int lane = tid & 31;

    for (int i = tid; i < Hdim; i += 256) hS[i] = h_in[i];
    __syncthreads();

    // 128 gate rows per block: r in [0,128), gate = r>>5, local unit = r&31
#pragma unroll 1
    for (int rr = 0; rr < 16; ++rr) {
        int r    = wave * 16 + rr;
        int gate = r >> 5;
        int unit = blockIdx.x * 32 + (r & 31);
        int row  = gate * Hdim + unit;
        const unsigned int* wp = (const unsigned int*)(w_hh + (size_t)row * Hdim);
        float acc = 0.f;
#pragma unroll
        for (int i = 0; i < 16; ++i) {          // 512 uints = 1024 bf16 per row
            unsigned int pk = wp[i * 32 + lane];
            int k = 2 * (i * 32 + lane);
            acc += bf2f(pk & 0xFFFFu) * hS[k] + bf2f(pk >> 16) * hS[k + 1];
        }
#pragma unroll
        for (int o = 16; o > 0; o >>= 1) acc += __shfl_down(acc, o);
        if (lane == 0) gS[r] = acc + xg[(size_t)t * Gdim + row];
    }
    __syncthreads();

    if (tid < 32) {
        int unit = blockIdx.x * 32 + tid;
        float iv = gS[tid];
        float fv = gS[32 + tid];
        float gv = gS[64 + tid];
        float ov = gS[96 + tid];
        float cn = sigm(fv) * c[unit] + sigm(iv) * tanhf(gv);
        float hn = sigm(ov) * tanhf(cn);
        c[unit] = cn;
        h_out[unit] = hn;
        if (out_f32) out_f32[(size_t)t * Hdim + unit] = hn;
        if (out_bf)  out_bf[(size_t)t * Hdim + unit]  = f2bf(hn);
    }
}

// ---------- attention ----------
// scores[s] = dot(enc[s,:], h2)   (one wave per score)
__global__ void scores_k(const float* __restrict__ enc, const float* __restrict__ h,
                         float* __restrict__ s) {
    int wid  = (blockIdx.x * blockDim.x + threadIdx.x) >> 5;
    int lane = threadIdx.x & 31;
    if (wid >= Sdim) return;
    const float* row = enc + (size_t)wid * Hdim;
    float acc = 0.f;
    for (int i = lane; i < Hdim; i += 32) acc += row[i] * h[i];
#pragma unroll
    for (int o = 16; o > 0; o >>= 1) acc += __shfl_down(acc, o);
    if (lane == 0) s[wid] = acc;
}

// softmax over 4096 in place (single block)
__global__ void softmax_k(float* __restrict__ s) {
    __shared__ float red[256];
    int tid = threadIdx.x;
    float m = -1e30f;
    for (int i = tid; i < Sdim; i += 256) m = fmaxf(m, s[i]);
    red[tid] = m; __syncthreads();
    for (int o = 128; o > 0; o >>= 1) { if (tid < o) red[tid] = fmaxf(red[tid], red[tid + o]); __syncthreads(); }
    m = red[0]; __syncthreads();
    float sum = 0.f;
    for (int i = tid; i < Sdim; i += 256) { float e = __expf(s[i] - m); s[i] = e; sum += e; }
    red[tid] = sum; __syncthreads();
    for (int o = 128; o > 0; o >>= 1) { if (tid < o) red[tid] += red[tid + o]; __syncthreads(); }
    float inv = 1.0f / red[0];
    __syncthreads();
    for (int i = tid; i < Sdim; i += 256) s[i] *= inv;
}

// context[h] = sum_s w[s] * enc[s,h]   (thread per h, coalesced over h)
__global__ void context_k(const float* __restrict__ enc, const float* __restrict__ w,
                          float* __restrict__ ctx) {
    int h = blockIdx.x * blockDim.x + threadIdx.x;
    if (h >= Hdim) return;
    float acc = 0.f;
    for (int s = 0; s < Sdim; ++s) acc += w[s] * enc[(size_t)s * Hdim + h];
    ctx[h] = acc;
}

// out[j] = tanh( dot(w_cat[j,:], [ctx ; h2]) + b_cat[j] )   (wave per j)
__global__ void attn_out_k(const float* __restrict__ w_cat, const float* __restrict__ b_cat,
                           const float* __restrict__ ctx, const float* __restrict__ h2,
                           float* __restrict__ out) {
    int wid  = (blockIdx.x * blockDim.x + threadIdx.x) >> 5;
    int lane = threadIdx.x & 31;
    if (wid >= Hdim) return;
    const float* row = w_cat + (size_t)wid * (2 * Hdim);
    float acc = 0.f;
    for (int i = lane; i < 2 * Hdim; i += 32) {
        float v = (i < Hdim) ? ctx[i] : h2[i - Hdim];
        acc += row[i] * v;
    }
#pragma unroll
    for (int o = 16; o > 0; o >>= 1) acc += __shfl_down(acc, o);
    if (lane == 0) out[wid] = tanhf(acc + b_cat[wid]);
}

// ---------- host orchestration ----------
extern "C" void kernel_launch(void* const* d_in, const int* in_sizes, int n_in,
                              void* d_out, int out_size, void* d_ws, size_t ws_size,
                              hipStream_t stream) {
    const float* x        = (const float*)d_in[0];   // (T,1,H)
    const float* enc      = (const float*)d_in[1];   // (S,1,H)
    const float* h0       = (const float*)d_in[2];   // (2,1,H)
    const float* c0       = (const float*)d_in[3];
    const float* w_ih0    = (const float*)d_in[4];   // (4H,H)
    const float* w_hh0    = (const float*)d_in[5];
    const float* b_ih0    = (const float*)d_in[6];
    const float* b_hh0    = (const float*)d_in[7];
    const float* w_ih1    = (const float*)d_in[8];
    const float* w_hh1    = (const float*)d_in[9];
    const float* b_ih1    = (const float*)d_in[10];
    const float* b_hh1    = (const float*)d_in[11];
    const float* w_cat    = (const float*)d_in[12];  // (H,2H)
    const float* b_cat    = (const float*)d_in[13];
    float* out = (float*)d_out;

    // workspace carve-up
    char* ws = (char*)d_ws;
    size_t off = 0;
    auto carve = [&](size_t bytes) { char* p = ws + off; off += (bytes + 255) & ~size_t(255); return p; };
    unsigned short* w_ih0_bf = (unsigned short*)carve((size_t)Gdim * Hdim * 2);
    unsigned short* w_hh0_bf = (unsigned short*)carve((size_t)Gdim * Hdim * 2);
    unsigned short* w_ih1_bf = (unsigned short*)carve((size_t)Gdim * Hdim * 2);
    unsigned short* w_hh1_bf = (unsigned short*)carve((size_t)Gdim * Hdim * 2);
    unsigned short* x_bf     = (unsigned short*)carve((size_t)Tdim * Hdim * 2);
    unsigned short* out0_bf  = (unsigned short*)carve((size_t)Tdim * Hdim * 2);
    float* xg0   = (float*)carve((size_t)Tdim * Gdim * 4);
    float* xg1   = (float*)carve((size_t)Tdim * Gdim * 4);
    float* hA    = (float*)carve(Hdim * 4);
    float* hB    = (float*)carve(Hdim * 4);
    float* cbuf  = (float*)carve(Hdim * 4);
    float* score = (float*)carve(Sdim * 4);
    float* ctx   = (float*)carve(Hdim * 4);

    const int WM = Gdim * Hdim;  // weight element count

    // 1) bf16 conversions
    cvt_bf16_k<<<4096, 256, 0, stream>>>(w_ih0, w_ih0_bf, WM);
    cvt_bf16_k<<<4096, 256, 0, stream>>>(w_hh0, w_hh0_bf, WM);
    cvt_bf16_k<<<4096, 256, 0, stream>>>(w_ih1, w_ih1_bf, WM);
    cvt_bf16_k<<<4096, 256, 0, stream>>>(w_hh1, w_hh1_bf, WM);
    cvt_bf16_k<<<2048, 256, 0, stream>>>(x, x_bf, Tdim * Hdim);

    // 2) xg0 = X @ w_ih0^T + (b_ih0 + b_hh0)   -- WMMA
    {
        int tiles = (Tdim / 16) * (Gdim / 16);   // 16384 tiles, 8 waves/block
        gemm_wmma_bf16_k<<<tiles / 8, 256, 0, stream>>>(x_bf, w_ih0_bf, b_ih0, b_hh0,
                                                        xg0, Tdim, Gdim, Hdim);
    }

    // 3) layer-0 recurrence
    copy_f32_k<<<4, 256, 0, stream>>>(hA, h0, Hdim);
    copy_f32_k<<<4, 256, 0, stream>>>(cbuf, c0, Hdim);
    for (int t = 0; t < Tdim; ++t) {
        const float* hin = (t & 1) ? hB : hA;
        float*       hout = (t & 1) ? hA : hB;
        lstm_step_k<<<32, 256, 0, stream>>>(xg0, w_hh0_bf, hin, hout, cbuf,
                                            nullptr, out0_bf, t);
    }
    // final layer-0 state (Tdim even -> lands in hA): hN[0], cN[0]
    copy_f32_k<<<4, 256, 0, stream>>>(out + (size_t)Tdim * Hdim,            hA,   Hdim);
    copy_f32_k<<<4, 256, 0, stream>>>(out + (size_t)Tdim * Hdim + 2 * Hdim, cbuf, Hdim);

    // 4) xg1 = out0 @ w_ih1^T + (b_ih1 + b_hh1)   -- WMMA
    {
        int tiles = (Tdim / 16) * (Gdim / 16);
        gemm_wmma_bf16_k<<<tiles / 8, 256, 0, stream>>>(out0_bf, w_ih1_bf, b_ih1, b_hh1,
                                                        xg1, Tdim, Gdim, Hdim);
    }

    // 5) layer-1 recurrence (writes out1 rows directly to d_out)
    copy_f32_k<<<4, 256, 0, stream>>>(hA, h0 + Hdim, Hdim);
    copy_f32_k<<<4, 256, 0, stream>>>(cbuf, c0 + Hdim, Hdim);
    for (int t = 0; t < Tdim; ++t) {
        const float* hin = (t & 1) ? hB : hA;
        float*       hout = (t & 1) ? hA : hB;
        lstm_step_k<<<32, 256, 0, stream>>>(xg1, w_hh1_bf, hin, hout, cbuf,
                                            out, nullptr, t);
    }
    // final layer-1 state: hN[1], cN[1]
    copy_f32_k<<<4, 256, 0, stream>>>(out + (size_t)Tdim * Hdim + Hdim,     hA,   Hdim);
    copy_f32_k<<<4, 256, 0, stream>>>(out + (size_t)Tdim * Hdim + 3 * Hdim, cbuf, Hdim);

    // 6) attention: scores -> softmax -> context -> tanh(cat @ w_cat^T + b)
    scores_k<<<Sdim / 8, 256, 0, stream>>>(enc, hA, score);
    softmax_k<<<1, 256, 0, stream>>>(score);
    context_k<<<Hdim / 256, 256, 0, stream>>>(enc, score, ctx);
    attn_out_k<<<Hdim / 8, 256, 0, stream>>>(w_cat, b_cat, ctx, hA,
                                             out + (size_t)Tdim * Hdim + 4 * Hdim);
}
